// Attention_59528246723275
// MI455X (gfx1250) — compile-verified, matrix operand order checked by reading, not run on hardware
//
#include <hip/hip_runtime.h>
#include <math.h>

// Problem constants (match reference)
#define Bv   2
#define Sv   2048
#define Hv   32
#define HKVv 8
#define Dv   64
#define HIDv 2048
#define NREP (Hv / HKVv)   // 4

// ---- CDNA5 WMMA types (wave32, 16x16x32 bf16 -> f32) ----
typedef __attribute__((ext_vector_type(16))) __bf16        v16bf;
typedef __attribute__((ext_vector_type(8)))  float         v8f;
typedef __attribute__((ext_vector_type(4)))  unsigned int  u32x4;

struct Frag {
  union { v16bf v; u32x4 q[2]; unsigned short u[16]; };
};

static __device__ __forceinline__ unsigned short f32_to_bf16(float f) {
  unsigned int u = __float_as_uint(f);
  u += 0x7fffu + ((u >> 16) & 1u);   // round-to-nearest-even
  return (unsigned short)(u >> 16);
}
static __device__ __forceinline__ float bf16_to_f32(unsigned short h) {
  return __uint_as_float(((unsigned int)h) << 16);
}

static __device__ __forceinline__ v8f wmma_bf16(const Frag& a, const Frag& b, v8f c) {
  // (neg_a, A, neg_b, B, c_mod, C, reuse_a, reuse_b)
  return __builtin_amdgcn_wmma_f32_16x16x32_bf16(false, a.v, false, b.v,
                                                 (short)0, c, false, false);
}

// 16-bit A/B fragment from a row-major (row, k) source:
// lane L: row = L%16; elems 0..7 -> k = kbase + 8*(L/16) + e
//                     elems 8..15 -> k = kbase + 16 + 8*(L/16) + (e-8)
static __device__ __forceinline__ void load_frag_row(Frag& f, const unsigned short* rowptr,
                                                     int kbase, int half) {
  f.q[0] = *(const u32x4*)(rowptr + kbase + (half << 3));
  f.q[1] = *(const u32x4*)(rowptr + kbase + 16 + (half << 3));
}

// ---- Async global -> LDS copy of 16 bytes per lane (GLOBAL_LOAD_ASYNC_TO_LDS_B128) ----
// VDST VGPR = 32-bit LDS byte address (addrspace(3) offset), VADDR = 64-bit global addr.
static __device__ __forceinline__ void async_copy16(const unsigned short* gsrc,
                                                    unsigned short* ldst) {
  unsigned lofs = (unsigned)(unsigned long long)
      (__attribute__((address_space(3))) const unsigned short*)ldst;
  unsigned long long ga = (unsigned long long)gsrc;
  asm volatile("global_load_async_to_lds_b128 %0, %1, off"
               :: "v"(lofs), "v"(ga) : "memory");
}

static __device__ __forceinline__ void wait_async0() {
  asm volatile("s_wait_asynccnt 0x0" ::: "memory");
}

// ---------------- f32 -> bf16 convert ----------------
__global__ __launch_bounds__(256) void cvt_f32_bf16(const float* __restrict__ in,
                                                    unsigned short* __restrict__ out, int n) {
  int i = blockIdx.x * blockDim.x + threadIdx.x;
  int stride = gridDim.x * blockDim.x;
  for (; i < n; i += stride) out[i] = f32_to_bf16(in[i]);
}

// ---------------- GEMM: C[m][n] = sum_k A[m][k] * W[n][k] ----------------
// A: (M,K) bf16 row-major, W: (N,K) bf16 row-major (i.e. C = A @ W^T).
// One wave computes a 16x64 tile (4 WMMA accumulators).
__global__ __launch_bounds__(256) void gemm_bf16_nt(const unsigned short* __restrict__ A,
                                                    const unsigned short* __restrict__ W,
                                                    void* __restrict__ Cout,
                                                    int M, int N, int K, int out_bf16) {
  const int lane = threadIdx.x & 31;
  const int wave = threadIdx.x >> 5;
  const int lm = lane & 15, half = lane >> 4;
  int flat = blockIdx.x * (blockDim.x >> 5) + wave;
  int mtiles = M >> 4, ntiles = N >> 6;
  if (flat >= mtiles * ntiles) return;
  int mt = flat % mtiles, nt = flat / mtiles;
  int m0 = mt << 4, n0 = nt << 6;

  v8f zero = {};
  v8f acc[4];
#pragma unroll
  for (int t = 0; t < 4; ++t) acc[t] = zero;

  const unsigned short* arow  = A + (size_t)(m0 + lm) * K;
  const unsigned short* wrow0 = W + (size_t)(n0 + lm) * K;
  for (int k0 = 0; k0 < K; k0 += 32) {
    // keep the HBM/L2 pipe primed one k-block ahead (global_prefetch_b8)
    if (k0 + 32 < K) {
      __builtin_prefetch(arow + k0 + 32 + (half << 3), 0, 1);
      __builtin_prefetch(wrow0 + k0 + 32 + (half << 3), 0, 1);
    }
    Frag af; load_frag_row(af, arow, k0, half);
#pragma unroll
    for (int t = 0; t < 4; ++t) {
      Frag wf; load_frag_row(wf, wrow0 + (size_t)(t * 16) * K, k0, half);
      acc[t] = wmma_bf16(af, wf, acc[t]);
    }
  }
  // C/D layout: lane L holds col n=L%16; VGPR v holds row m = v + 8*(L/16)
#pragma unroll
  for (int t = 0; t < 4; ++t) {
#pragma unroll
    for (int v = 0; v < 8; ++v) {
      size_t off = (size_t)(m0 + v + 8 * half) * N + (size_t)(n0 + 16 * t + lm);
      float val = acc[t][v];
      if (out_bf16) ((unsigned short*)Cout)[off] = f32_to_bf16(val);
      else          ((float*)Cout)[off] = val;
    }
  }
}

// ---------------- RoPE, in place on bf16 (B,S,nheads,D) ----------------
__global__ __launch_bounds__(256) void rope_bf16(unsigned short* __restrict__ X,
                                                 const float* __restrict__ cosb,
                                                 const float* __restrict__ sinb, int nheads) {
  int idx = blockIdx.x * blockDim.x + threadIdx.x;
  int total = Bv * Sv * nheads;
  if (idx >= total) return;
  int s = (idx / nheads) % Sv;
  unsigned short* row = X + (size_t)idx * Dv;
  const float* c  = cosb + (size_t)s * Dv;
  const float* sn = sinb + (size_t)s * Dv;
  float q[Dv];
#pragma unroll
  for (int d = 0; d < Dv; ++d) q[d] = bf16_to_f32(row[d]);
#pragma unroll
  for (int d = 0; d < Dv; ++d) {
    float rot = (d < Dv / 2) ? -q[d + Dv / 2] : q[d - Dv / 2];
    row[d] = f32_to_bf16(q[d] * c[d] + rot * sn[d]);
  }
}

// ---------------- Flash attention (causal, GQA), one wave per 16-row Q tile ----
// Q: (B,S,H,D) bf16 (RoPE'd)  K,V: (B,S,HKV,D) bf16  O: (B,S,H,D) bf16
// K/V 32-key tiles are staged into LDS with GLOBAL_LOAD_ASYNC_TO_LDS_B128.
__global__ __launch_bounds__(256) void attn_fwd(const unsigned short* __restrict__ Q,
                                                const unsigned short* __restrict__ Kc,
                                                const unsigned short* __restrict__ Vc,
                                                unsigned short* __restrict__ O) {
  __shared__ unsigned short lds_k[8][32 * Dv];   // 4 KB per wave
  __shared__ unsigned short lds_v[8][32 * Dv];   // 4 KB per wave
  __shared__ unsigned short lds_p[8][16 * 32];   // 1 KB per wave
  const int lane = threadIdx.x & 31;
  const int wave = threadIdx.x >> 5;
  const int lm = lane & 15, half = lane >> 4;
  int flat = blockIdx.x * (blockDim.x >> 5) + wave;
  const int qtiles = Sv / 16;
  int qt = flat % qtiles;
  int hb = flat / qtiles;
  int h  = hb % Hv;
  int b  = hb / Hv;
  int q0 = qt * 16;
  int hkv = h / NREP;

  // Q fragments (rows q0..q0+15, full D=64 as two k=32 fragments)
  const unsigned short* qrow = Q + ((size_t)((b * Sv + q0 + lm) * Hv + h)) * Dv;
  Frag qf0, qf1;
  load_frag_row(qf0, qrow, 0,  half);
  load_frag_row(qf1, qrow, 32, half);

  v8f zero = {};
  v8f o[4];
#pragma unroll
  for (int t = 0; t < 4; ++t) o[t] = zero;
  float mrow[8], lrow[8];
#pragma unroll
  for (int v = 0; v < 8; ++v) { mrow[v] = -3.0e38f; lrow[v] = 0.f; }

  const unsigned short* kbase = Kc + ((size_t)(b * Sv) * HKVv + hkv) * Dv;
  const unsigned short* vbase = Vc + ((size_t)(b * Sv) * HKVv + hkv) * Dv;
  unsigned short* lk = lds_k[wave];
  unsigned short* lv = lds_v[wave];
  unsigned short* pl = lds_p[wave];
  const int krowstride = HKVv * Dv;   // elements between consecutive keys

  for (int k0 = 0; k0 <= q0 + 15; k0 += 32) {
    // ---- stage 32-key K/V tiles into LDS (async, 16B per lane per op) ----
    // tile = 32 rows x 64 bf16 = 4 KB = 256 16B-chunks; chunk c: row=c/8, off=(c%8)*8 elems
#pragma unroll
    for (int i = 0; i < 8; ++i) {
      int c   = i * 32 + lane;
      int row = c >> 3;
      int cof = (c & 7) << 3;
      const unsigned short* gk = kbase + (size_t)(k0 + row) * krowstride + cof;
      const unsigned short* gv = vbase + (size_t)(k0 + row) * krowstride + cof;
      async_copy16(gk, lk + row * Dv + cof);
      async_copy16(gv, lv + row * Dv + cof);
    }
    wait_async0();

    // ---- scores: S0 (keys k0..k0+15), S1 (keys k0+16..k0+31) from LDS ----
    v8f s0 = zero, s1 = zero;
    Frag kf;
    load_frag_row(kf, lk + (size_t)lm * Dv, 0,  half);        s0 = wmma_bf16(qf0, kf, s0);
    load_frag_row(kf, lk + (size_t)lm * Dv, 32, half);        s0 = wmma_bf16(qf1, kf, s0);
    load_frag_row(kf, lk + (size_t)(16 + lm) * Dv, 0,  half); s1 = wmma_bf16(qf0, kf, s1);
    load_frag_row(kf, lk + (size_t)(16 + lm) * Dv, 32, half); s1 = wmma_bf16(qf1, kf, s1);

    // ---- scale + causal mask (1/sqrt(64) = 0.125) ----
    float p0[8], p1[8];
#pragma unroll
    for (int v = 0; v < 8; ++v) {
      int qi = q0 + v + 8 * half;
      float a  = s0[v] * 0.125f;
      float bb = s1[v] * 0.125f;
      if (k0 + lm > qi)      a  = -3.0e38f;
      if (k0 + 16 + lm > qi) bb = -3.0e38f;
      p0[v] = a; p1[v] = bb;
    }
    // ---- online softmax per row (rows live across 16-lane halves) ----
#pragma unroll
    for (int v = 0; v < 8; ++v) {
      float mx = fmaxf(p0[v], p1[v]);
#pragma unroll
      for (int msk = 1; msk < 16; msk <<= 1) mx = fmaxf(mx, __shfl_xor(mx, msk, 32));
      float mnew = fmaxf(mrow[v], mx);
      float fac  = __expf(mrow[v] - mnew);
      mrow[v] = mnew;
      p0[v] = __expf(p0[v] - mnew);
      p1[v] = __expf(p1[v] - mnew);
      float rs = p0[v] + p1[v];
#pragma unroll
      for (int msk = 1; msk < 16; msk <<= 1) rs += __shfl_xor(rs, msk, 32);
      lrow[v] = lrow[v] * fac + rs;
#pragma unroll
      for (int t = 0; t < 4; ++t) o[t][v] *= fac;
    }
    // ---- P (C/D layout) -> LDS (row-major 16x32) -> A fragment ----
#pragma unroll
    for (int v = 0; v < 8; ++v) {
      pl[(v + 8 * half) * 32 + lm]      = f32_to_bf16(p0[v]);
      pl[(v + 8 * half) * 32 + 16 + lm] = f32_to_bf16(p1[v]);
    }
    asm volatile("s_wait_dscnt 0" ::: "memory");   // wave-internal LDS RAW
    Frag pf;
    pf.q[0] = *(const u32x4*)(pl + lm * 32 + (half << 3));
    pf.q[1] = *(const u32x4*)(pl + lm * 32 + 16 + (half << 3));
    // ---- P @ V from LDS (B fragment: n = d column, contraction = keys) ----
#pragma unroll
    for (int t = 0; t < 4; ++t) {
      Frag vf;
#pragma unroll
      for (int e = 0; e < 16; ++e) {
        int kk = (e < 8) ? (8 * half + e) : (16 + 8 * half + (e - 8));
        vf.u[e] = lv[kk * Dv + 16 * t + lm];
      }
      o[t] = wmma_bf16(pf, vf, o[t]);
    }
  }
  // ---- normalize and store O (bf16, (B,S,H,D)) ----
#pragma unroll
  for (int v = 0; v < 8; ++v) {
    float inv = 1.0f / lrow[v];
    int qi = q0 + v + 8 * half;
    size_t base = ((size_t)((b * Sv + qi) * Hv + h)) * Dv;
#pragma unroll
    for (int t = 0; t < 4; ++t)
      O[base + 16 * t + lm] = f32_to_bf16(o[t][v] * inv);
  }
}

// ---------------- host-side launch ----------------
extern "C" void kernel_launch(void* const* d_in, const int* in_sizes, int n_in,
                              void* d_out, int out_size, void* d_ws, size_t ws_size,
                              hipStream_t stream) {
  (void)in_sizes; (void)n_in; (void)out_size; (void)ws_size;
  const float* hs   = (const float*)d_in[0];
  const float* cosb = (const float*)d_in[1];
  const float* sinb = (const float*)d_in[2];
  const float* wq   = (const float*)d_in[3];
  const float* wk   = (const float*)d_in[4];
  const float* wv   = (const float*)d_in[5];
  const float* wo   = (const float*)d_in[6];
  float* out = (float*)d_out;

  const size_t M = (size_t)Bv * Sv;   // 4096 token rows
  char* ws = (char*)d_ws;
  size_t off = 0;
  auto alloc = [&](size_t bytes) -> char* {
    char* p = ws + off;
    off += (bytes + 255) & ~(size_t)255;
    return p;
  };
  unsigned short* hs_b = (unsigned short*)alloc(M * HIDv * 2);
  unsigned short* wq_b = (unsigned short*)alloc((size_t)Hv * Dv * HIDv * 2);
  unsigned short* wk_b = (unsigned short*)alloc((size_t)HKVv * Dv * HIDv * 2);
  unsigned short* wv_b = (unsigned short*)alloc((size_t)HKVv * Dv * HIDv * 2);
  unsigned short* wo_b = (unsigned short*)alloc((size_t)HIDv * Hv * Dv * 2);
  unsigned short* Qb   = (unsigned short*)alloc(M * Hv * Dv * 2);
  unsigned short* Kb   = (unsigned short*)alloc(M * HKVv * Dv * 2);
  unsigned short* Vb   = (unsigned short*)alloc(M * HKVv * Dv * 2);
  unsigned short* Ob   = (unsigned short*)alloc(M * Hv * Dv * 2);
  // total ~80 MB of d_ws

  // 1) f32 -> bf16 converts
  cvt_f32_bf16<<<2048, 256, 0, stream>>>(hs, hs_b, (int)(M * HIDv));
  cvt_f32_bf16<<<2048, 256, 0, stream>>>(wq, wq_b, Hv * Dv * HIDv);
  cvt_f32_bf16<<<1024, 256, 0, stream>>>(wk, wk_b, HKVv * Dv * HIDv);
  cvt_f32_bf16<<<1024, 256, 0, stream>>>(wv, wv_b, HKVv * Dv * HIDv);
  cvt_f32_bf16<<<2048, 256, 0, stream>>>(wo, wo_b, HIDv * Hv * Dv);

  // 2) QKV projections (bf16 epilogue)
  {
    int tiles = (int)(M / 16) * ((Hv * Dv) / 64);
    gemm_bf16_nt<<<(tiles + 7) / 8, 256, 0, stream>>>(hs_b, wq_b, Qb, (int)M, Hv * Dv, HIDv, 1);
  }
  {
    int tiles = (int)(M / 16) * ((HKVv * Dv) / 64);
    gemm_bf16_nt<<<(tiles + 7) / 8, 256, 0, stream>>>(hs_b, wk_b, Kb, (int)M, HKVv * Dv, HIDv, 1);
    gemm_bf16_nt<<<(tiles + 7) / 8, 256, 0, stream>>>(hs_b, wv_b, Vb, (int)M, HKVv * Dv, HIDv, 1);
  }

  // 3) RoPE (in place)
  rope_bf16<<<((Bv * Sv * Hv) + 255) / 256, 256, 0, stream>>>(Qb, cosb, sinb, Hv);
  rope_bf16<<<((Bv * Sv * HKVv) + 255) / 256, 256, 0, stream>>>(Kb, cosb, sinb, HKVv);

  // 4) Flash attention: one wave per (b, h, 16-row q tile) = 8192 waves
  {
    int waves = Bv * Hv * (Sv / 16);
    attn_fwd<<<waves / 8, 256, 0, stream>>>(Qb, Kb, Vb, Ob);
  }

  // 5) Output projection (f32 epilogue into d_out)
  {
    int tiles = (int)(M / 16) * (HIDv / 64);
    gemm_bf16_nt<<<(tiles + 7) / 8, 256, 0, stream>>>(Ob, wo_b, out, (int)M, HIDv, Hv * Dv, 0);
  }
}
